// LinearCritic_73615739453903
// MI455X (gfx1250) — matrix operand order repaired and checked by reading, not compiled, not run on hardware
//
#include <hip/hip_runtime.h>
#include <stdint.h>

typedef __attribute__((ext_vector_type(16))) __bf16 v16bf;
typedef __attribute__((ext_vector_type(8)))  float  v8f;

#define TEMPERATURE 0.5f
#define BN_EPS 1e-5f
#define COS_EPS 1e-8f

#define BM 128
#define BN 128
#define BK 32
#define LDT 48    // LDS tile row stride in bf16 elements (96B, multiple of 16B)
#define SSTR 132  // epilogue staging row stride in floats (528B, 16B aligned)

#define USE_ASYNC 1   // global_load_async_to_lds_b128 path; set 0 to fall back

static __device__ __forceinline__ unsigned short f2bf(float f) {
  union { float f; unsigned int u; } x; x.f = f;
  unsigned int u = x.u;
  u += 0x7FFFu + ((u >> 16) & 1u);   // round-to-nearest-even
  return (unsigned short)(u >> 16);
}

// ---------------- f32 -> bf16 conversion (x4 vectorized) ----------------
__global__ void k_cvt4(const float4* __restrict__ src, ushort4* __restrict__ dst, long n4) {
  long i = (long)blockIdx.x * blockDim.x + threadIdx.x;
  long stride = (long)gridDim.x * blockDim.x;
  for (; i < n4; i += stride) {
    float4 v = src[i];
    ushort4 o;
    o.x = f2bf(v.x); o.y = f2bf(v.y); o.z = f2bf(v.z); o.w = f2bf(v.w);
    dst[i] = o;
  }
}

// ---------------- tile staging: global (row-major, K contiguous) -> LDS ----------------
static __device__ __forceinline__ void tile_load(const unsigned short* __restrict__ g, int ld,
                                                 int rowBase, int kb,
                                                 unsigned short* __restrict__ lds, int tid) {
  int row = tid >> 1;            // 0..127
  int seg = (tid & 1) << 4;      // 0 or 16 elements
  const unsigned short* src = g + (long)(rowBase + row) * ld + kb + seg;
  unsigned short* dst = lds + row * LDT + seg;
#if USE_ASYNC
  unsigned int d0 = (unsigned int)(unsigned long long)dst;        // LDS byte offset (low 32 bits)
  asm volatile("global_load_async_to_lds_b128 %0, %1, off"
               :: "v"(d0), "v"(src) : "memory");
  asm volatile("global_load_async_to_lds_b128 %0, %1, off"
               :: "v"(d0 + 16u), "v"(src + 8) : "memory");
#else
  uint4 a = *(const uint4*)src;
  uint4 b = *(const uint4*)(src + 8);
  uint4* d = (uint4*)dst;
  d[0] = a; d[1] = b;
#endif
}

static __device__ __forceinline__ void tiles_wait() {
#if USE_ASYNC
  asm volatile("s_wait_asynccnt 0x0" ::: "memory");
#endif
}

// A 16x32 bf16 fragment: lanes 0-15 row M=lane hold K {0..7,16..23}; lanes 16-31 K {8..15,24..31}
static __device__ __forceinline__ v16bf load_afrag(const unsigned short* __restrict__ lds,
                                                   int mt, int lane) {
  int m = lane & 15;
  int kh = (lane >> 4) << 3;     // 0 or 8
  const unsigned short* p = lds + (mt + m) * LDT + kh;
  v16bf r;
  ((uint4*)&r)[0] = *(const uint4*)p;         // K kh..kh+7
  ((uint4*)&r)[1] = *(const uint4*)(p + 16);  // K 16+kh..16+kh+7
  return r;
}

// B 32x16 bf16 fragment: lane holds column N=lane%16; lanes 0-15 K {0..7,16..23}; lanes 16-31 K {8..15,24..31}
static __device__ __forceinline__ v16bf load_bfrag(const unsigned short* __restrict__ lds,
                                                   int nt, int lane) {
  int n = lane & 15;
  int ks = (lane >> 4) << 3;     // 0 or 8
  const unsigned short* p = lds + (nt + n) * LDT + ks;
  v16bf r;
  ((uint4*)&r)[0] = *(const uint4*)p;
  ((uint4*)&r)[1] = *(const uint4*)(p + 16);
  return r;
}

// ---------------- bf16 WMMA GEMM: C[M,N] = A[M,K] @ B[N,K]^T ----------------
// SIM: A==B==P, C scaled by 1/max(n_i n_j, eps)/T with main diagonal = -inf.
template<bool SIM>
__global__ __launch_bounds__(256) void k_mm(const unsigned short* __restrict__ A,
                                            const unsigned short* __restrict__ B,
                                            float* __restrict__ C,
                                            const float* __restrict__ nrm,
                                            int Ndim, int K) {
  __shared__ __align__(16) unsigned short tiles[2][(BM + BN) * LDT];  // 48 KB ping-pong
  int tid  = threadIdx.x;
  int lane = tid & 31;
  int wave = tid >> 5;
  int wm = (wave & 3) * 32;   // 4 waves along M
  int wn = (wave >> 2) * 64;  // 2 waves along N
  int row0 = blockIdx.y * BM;
  int col0 = blockIdx.x * BN;

  v8f acc[2][4];
  #pragma unroll
  for (int i = 0; i < 2; ++i)
    #pragma unroll
    for (int j = 0; j < 4; ++j)
      #pragma unroll
      for (int g = 0; g < 8; ++g) acc[i][j][g] = 0.0f;

  int nk = K / BK;
  // prologue: stage first K-slice into buffer 0
  tile_load(A, K, row0, 0, &tiles[0][0], tid);
  tile_load(B, K, col0, 0, &tiles[0][BM * LDT], tid);

  for (int kt = 0; kt < nk; ++kt) {
    tiles_wait();          // wave's own async copies done
    __syncthreads();       // all waves' copies visible
    int cur = kt & 1;
    if (kt + 1 < nk) {     // overlap next slice's DMA with this slice's WMMAs
      tile_load(A, K, row0, (kt + 1) * BK, &tiles[cur ^ 1][0], tid);
      tile_load(B, K, col0, (kt + 1) * BK, &tiles[cur ^ 1][BM * LDT], tid);
    }
    const unsigned short* As = &tiles[cur][0];
    const unsigned short* Bs = &tiles[cur][BM * LDT];
    v16bf af[2], bfm[4];
    #pragma unroll
    for (int i = 0; i < 2; ++i) af[i] = load_afrag(As, wm + i * 16, lane);
    #pragma unroll
    for (int j = 0; j < 4; ++j) bfm[j] = load_bfrag(Bs, wn + j * 16, lane);
    #pragma unroll
    for (int i = 0; i < 2; ++i)
      #pragma unroll
      for (int j = 0; j < 4; ++j)
        acc[i][j] = __builtin_amdgcn_wmma_f32_16x16x32_bf16(
            false, af[i], false, bfm[j], (short)0, acc[i][j], false, false);
  }

  // ---------------- staged, coalesced epilogue ----------------
  __syncthreads();                      // tile buffers dead; reuse as f32 staging
  float* stage = (float*)&tiles[0][0];  // 32 x SSTR floats (16.9 KB)
  const float NEG_INF = -__builtin_huge_valf();
  int n   = lane & 15;
  int mbh = (lane >> 4) * 8;            // D layout: VGPR g -> M = g + 8*(lane/16)
  for (int chunk = 0; chunk < 4; ++chunk) {
    if (wm == chunk * 32) {             // the 2 waves owning these 32 rows stage their accs
      #pragma unroll
      for (int i = 0; i < 2; ++i)
        #pragma unroll
        for (int j = 0; j < 4; ++j) {
          int rloc = i * 16 + mbh;
          #pragma unroll
          for (int g = 0; g < 8; ++g)
            stage[(rloc + g) * SSTR + wn + j * 16 + n] = acc[i][j][g];
        }
    }
    __syncthreads();
    // all 256 threads copy 32x128 floats out, fully coalesced b128 stores
    int rr = tid >> 3;                  // 0..31
    int cc = (tid & 7) * 16;            // 0..112
    int gr = row0 + chunk * 32 + rr;
    float rn = SIM ? nrm[gr] : 0.0f;
    #pragma unroll
    for (int k = 0; k < 4; ++k) {
      float4 v = *(const float4*)&stage[rr * SSTR + cc + k * 4];
      int gc = col0 + cc + k * 4;
      if (SIM) {
        float4 nv = *(const float4*)&nrm[gc];
        v.x = v.x / fmaxf(rn * nv.x, COS_EPS) / TEMPERATURE;
        v.y = v.y / fmaxf(rn * nv.y, COS_EPS) / TEMPERATURE;
        v.z = v.z / fmaxf(rn * nv.z, COS_EPS) / TEMPERATURE;
        v.w = v.w / fmaxf(rn * nv.w, COS_EPS) / TEMPERATURE;
        if (gr == gc)     v.x = NEG_INF;
        if (gr == gc + 1) v.y = NEG_INF;
        if (gr == gc + 2) v.z = NEG_INF;
        if (gr == gc + 3) v.w = NEG_INF;
      }
      *(float4*)&C[(long)gr * Ndim + gc] = v;
    }
    __syncthreads();
  }
}

// ---------------- deterministic per-column batch stats (two-stage) ----------------
__global__ __launch_bounds__(256) void k_colstats_partial(const float* __restrict__ X,
                                                          float* __restrict__ part,
                                                          int ncols, int rowsPerHalf, int nchunks) {
  int c = blockIdx.x * 64 + (threadIdx.x & 63);
  int half = blockIdx.y, chunk = blockIdx.z;
  int rowsPerChunk = rowsPerHalf / nchunks;
  int r0 = half * rowsPerHalf + chunk * rowsPerChunk;
  int rt = threadIdx.x >> 6;   // 0..3
  float s = 0.f, q = 0.f;
  for (int r = rt; r < rowsPerChunk; r += 4) {
    float v = X[(long)(r0 + r) * ncols + c];
    s += v; q += v * v;
  }
  __shared__ float ls[256], lq[256];
  ls[threadIdx.x] = s; lq[threadIdx.x] = q;
  __syncthreads();
  if (threadIdx.x < 64) {
    float ss = ls[threadIdx.x] + ls[threadIdx.x + 64] + ls[threadIdx.x + 128] + ls[threadIdx.x + 192];
    float qq = lq[threadIdx.x] + lq[threadIdx.x + 64] + lq[threadIdx.x + 128] + lq[threadIdx.x + 192];
    long idx = (((long)half * nchunks + chunk) * ncols + c) * 2;
    part[idx] = ss; part[idx + 1] = qq;
  }
}

__global__ void k_colstats_final(const float* __restrict__ part, float* __restrict__ stats,
                                 int ncols, int rowsPerHalf, int nchunks) {
  int i = blockIdx.x * blockDim.x + threadIdx.x;   // over 2*ncols
  if (i >= 2 * ncols) return;
  int half = i / ncols, c = i % ncols;
  float s = 0.f, q = 0.f;
  for (int ch = 0; ch < nchunks; ++ch) {
    long idx = (((long)half * nchunks + ch) * ncols + c) * 2;
    s += part[idx]; q += part[idx + 1];
  }
  float mean = s / (float)rowsPerHalf;
  float var  = q / (float)rowsPerHalf - mean * mean;
  stats[(long)i * 2]     = mean;
  stats[(long)i * 2 + 1] = rsqrtf(var + BN_EPS);
}

// ---------------- BN1 + ReLU + cvt to bf16 ----------------
__global__ void k_bn_relu_cvt(const float* __restrict__ X, const float* __restrict__ stats,
                              const float* __restrict__ gamma, const float* __restrict__ beta,
                              unsigned short* __restrict__ H, int ncols, int rowsPerHalf) {
  long i = (long)blockIdx.x * blockDim.x + threadIdx.x;
  long n = (long)2 * rowsPerHalf * ncols;
  if (i >= n) return;
  int c = (int)(i % ncols);
  int r = (int)(i / ncols);
  int half = r / rowsPerHalf;
  const float* st = stats + ((long)half * ncols + c) * 2;
  float v = (X[i] - st[0]) * st[1] * gamma[c] + beta[c];
  v = v > 0.f ? v : 0.f;
  H[i] = f2bf(v);
}

// ---------------- BN2 (no affine) + row norms + cvt to bf16 ----------------
__global__ __launch_bounds__(256) void k_pnorm(const float* __restrict__ Y,
                                               const float* __restrict__ stats,
                                               unsigned short* __restrict__ Pb,
                                               float* __restrict__ nrm, int rowsPerHalf) {
  int t = threadIdx.x;
  int c = t & 127;
  int r = blockIdx.x * 2 + (t >> 7);
  int half = r / rowsPerHalf;
  const float* st = stats + ((long)half * 128 + c) * 2;
  float v = (Y[(long)r * 128 + c] - st[0]) * st[1];
  Pb[(long)r * 128 + c] = f2bf(v);
  __shared__ float ls[256];
  ls[t] = v * v;
  __syncthreads();
  for (int o = 64; o > 0; o >>= 1) {
    if ((t & 127) < o) ls[t] += ls[t + o];
    __syncthreads();
  }
  if ((t & 127) == 0) nrm[r] = sqrtf(ls[t]);
}

__global__ void k_targets(int* __restrict__ tgt, int n) {
  int i = blockIdx.x * blockDim.x + threadIdx.x;
  if (i < n) tgt[i] = i;
}

static inline size_t align256(size_t x) { return (x + 255) & ~(size_t)255; }

extern "C" void kernel_launch(void* const* d_in, const int* in_sizes, int n_in,
                              void* d_out, int out_size, void* d_ws, size_t ws_size,
                              hipStream_t stream) {
  const float* z1     = (const float*)d_in[0];
  const float* z2     = (const float*)d_in[1];
  const float* W1     = (const float*)d_in[2];
  const float* gamma1 = (const float*)d_in[3];
  const float* beta1  = (const float*)d_in[4];
  const float* W2     = (const float*)d_in[5];

  const int N = 4096, D = 2048, P = 128, M = 8192;
  const int NCHUNK = 16;

  char* ws = (char*)d_ws;
  size_t off = 0;
  unsigned short* Zb  = (unsigned short*)(ws + off); off += align256((size_t)M * D * 2); // reused as Hb
  unsigned short* W1b = (unsigned short*)(ws + off); off += align256((size_t)D * D * 2);
  unsigned short* W2b = (unsigned short*)(ws + off); off += align256((size_t)P * D * 2);
  float* X      = (float*)(ws + off); off += align256((size_t)M * D * 4);
  float* Y      = (float*)(ws + off); off += align256((size_t)M * P * 4);
  unsigned short* Pb = (unsigned short*)(ws + off); off += align256((size_t)M * P * 2);
  float* part   = (float*)(ws + off); off += align256((size_t)2 * NCHUNK * D * 2 * 4);
  float* stats1 = (float*)(ws + off); off += align256((size_t)2 * D * 2 * 4);
  float* stats2 = (float*)(ws + off); off += align256((size_t)2 * P * 2 * 4);
  float* nrm    = (float*)(ws + off); off += align256((size_t)M * 4);

  // 1. convert inputs to bf16
  k_cvt4<<<2048, 256, 0, stream>>>((const float4*)z1, (ushort4*)Zb, (long)N * D / 4);
  k_cvt4<<<2048, 256, 0, stream>>>((const float4*)z2, (ushort4*)(Zb + (size_t)N * D), (long)N * D / 4);
  k_cvt4<<<1024, 256, 0, stream>>>((const float4*)W1, (ushort4*)W1b, (long)D * D / 4);
  k_cvt4<<<256,  256, 0, stream>>>((const float4*)W2, (ushort4*)W2b, (long)P * D / 4);

  // 2. X = [z1;z2] @ W1^T   (8192 x 2048, K=2048)
  k_mm<false><<<dim3(D / BN, M / BM), 256, 0, stream>>>(Zb, W1b, X, nullptr, D, D);

  // 3. BN1 stats (per half) + BN1+ReLU into bf16 H (reuse Zb)
  k_colstats_partial<<<dim3(D / 64, 2, NCHUNK), 256, 0, stream>>>(X, part, D, N, NCHUNK);
  k_colstats_final<<<(2 * D + 255) / 256, 256, 0, stream>>>(part, stats1, D, N, NCHUNK);
  {
    long n = (long)M * D;
    k_bn_relu_cvt<<<(unsigned)((n + 255) / 256), 256, 0, stream>>>(X, stats1, gamma1, beta1, Zb, D, N);
  }

  // 4. Y = H @ W2^T   (8192 x 128, K=2048)
  k_mm<false><<<dim3(P / BN, M / BM), 256, 0, stream>>>(Zb, W2b, Y, nullptr, P, D);

  // 5. BN2 stats + normalize + row norms -> bf16 P
  k_colstats_partial<<<dim3(P / 64, 2, NCHUNK), 256, 0, stream>>>(Y, part, P, N, NCHUNK);
  k_colstats_final<<<1, 256, 0, stream>>>(part, stats2, P, N, NCHUNK);
  k_pnorm<<<M / 2, 256, 0, stream>>>(Y, stats2, Pb, nrm, N);

  // 6. similarity scores -> d_out (8192 x 8192 f32)
  k_mm<true><<<dim3(M / BN, M / BM), 256, 0, stream>>>(Pb, Pb, (float*)d_out, nrm, M, P);

  // 7. targets (int32 bit pattern) after the scores
  int* tgt = (int*)((float*)d_out + (size_t)M * M);
  k_targets<<<M / 256, 256, 0, stream>>>(tgt, M);

  (void)in_sizes; (void)n_in; (void)out_size; (void)ws_size;
}
